// SnseFilter_51436528337757
// MI455X (gfx1250) — matrix operand order repaired and checked by reading, not compiled, not run on hardware
//
#include <hip/hip_runtime.h>
#include <hip/hip_bf16.h>

// ---------------------------------------------------------------------------
// SNSE nonlinear filter for MI455X (gfx1250, wave32, WMMA).
//
// Dominant cost: 35 real 201-tap valid cross-correlations per (b,pol)
// sequence (~18 GFLOP total).  Mapped to v_wmma_f32_16x16x32_f16 via a
// Toeplitz-band formulation:
//   D[i,j] = sum_k Toep(w)[i,k] * stream_j[n0+32c+k],  7 K-chunks / tile
// => ~90% MAC efficiency, f32 accumulation, epilogue fused in-register
// using the documented 16x16 f32 C/D lane layout.
// ---------------------------------------------------------------------------

typedef __attribute__((ext_vector_type(16))) _Float16 v16h;
typedef __attribute__((ext_vector_type(8)))  float    v8f;

#define NB     16
#define MM     40000
#define NTAPS  201
#define TT     100          // NTAPS/2
#define MOUT   39800        // MM - NTAPS + 1
#define NSEQ   32           // B * Nm sequences, seq = b*2 + c
#define MPAD   40032        // padded stream length (>= 39792+224), 16-aligned
#define NST    19           // ps, qr, qi, {Ar,Ai,Br,Bi} x 4 shifts
#define NFILT  14           // ixpm, icixpm, {wr, wi, -wi} x 4 shifts
#define NCHUNK 7            // ceil(216/32) K-chunks per 16-output tile
#define NTILE  2488         // ceil(MOUT/16)

// workspace layout (bytes); total ~59 MB
#define WS_X_OFF  0ull                                          // float2 X[NSEQ][MM]
#define WS_T_OFF  (WS_X_OFF + (size_t)NSEQ * MM * 8)            // f16 T[NFILT][NCHUNK][32][16]
#define WS_S_OFF  (WS_T_OFF + (size_t)NFILT * NCHUNK * 32 * 16 * 2) // f16 S[NST][NSEQ][MPAD]
#define SSTRIDE   ((size_t)NSEQ * MPAD)

// ---------------------------------------------------------------------------
// Kernel 1: elementwise prep.  Scales x by sqrt(P), builds the 19 f16 conv
// input streams (with jnp.roll wraparound over M and pol-swap over Nm) and
// stores scaled complex x as float2 for the epilogue.
// ---------------------------------------------------------------------------
__global__ void prep_streams(const float* __restrict__ xr,
                             const float* __restrict__ xi,
                             const float* __restrict__ ti,
                             char* __restrict__ ws)
{
    int idx = blockIdx.x * blockDim.x + threadIdx.x;   // NSEQ*MPAD threads
    int seq = idx / MPAD;
    int m   = idx - seq * MPAD;

    _Float16* S = (_Float16*)(ws + WS_S_OFF);
    size_t sb = (size_t)seq * MPAD + m;

    if (m >= MM) {                                     // zero tail padding
        #pragma unroll
        for (int st = 0; st < NST; ++st) S[(size_t)st * SSTRIDE + sb] = (_Float16)0.f;
        return;
    }

    int b = seq >> 1, c = seq & 1;
    float p10 = __expf(ti[b * 4] * 0.23025850929940457f);  // 10^(ti/10)
    float sp  = sqrtf(p10 * 0.5f);                          // sqrt(P), P = 10^(ti/10)/Nm

    // scaled x at m-2..m+2 (mod M), both channels
    float cr[5], ci_[5], pr[5], pi[5];
    #pragma unroll
    for (int d = 0; d < 5; ++d) {
        int mm = m + d - 2;
        if (mm < 0) mm += MM;
        if (mm >= MM) mm -= MM;
        size_t base = ((size_t)b * MM + mm) * 2;
        cr[d] = xr[base + c]       * sp;  ci_[d] = xi[base + c]       * sp;
        pr[d] = xr[base + (1 - c)] * sp;  pi[d]  = xi[base + (1 - c)] * sp;
    }
    float xcr = cr[2], xci = ci_[2], xor_ = pr[2], xoi = pi[2];

    ((float2*)(ws + WS_X_OFF))[(size_t)seq * MM + m] = make_float2(xcr, xci);

    float pwc = xcr * xcr + xci * xci;
    float pwo = xor_ * xor_ + xoi * xoi;
    S[0 * SSTRIDE + sb] = (_Float16)(2.f * pwc + pwo);                 // ps
    S[1 * SSTRIDE + sb] = (_Float16)(xcr * xor_ + xci * xoi);          // q.real = x_c * conj(x_o)
    S[2 * SSTRIDE + sb] = (_Float16)(xci * xor_ - xcr * xoi);          // q.imag

    // FWM products: shift s in {-2,-1,1,2} -> x[m-s] is position index d = 2 - s
    const int dmap[4] = {4, 3, 1, 0};
    #pragma unroll
    for (int si = 0; si < 4; ++si) {
        int d = dmap[si];
        // A = 2*x_c*conj(xs_c) + x_o*conj(xs_o) ; B = x_c*conj(xs_o)
        float ar = 2.f * (xcr * cr[d] + xci * ci_[d]) + (xor_ * pr[d] + xoi * pi[d]);
        float ai = 2.f * (xci * cr[d] - xcr * ci_[d]) + (xoi * pr[d] - xor_ * pi[d]);
        float br = xcr * pr[d] + xci * pi[d];
        float bi = xci * pr[d] - xcr * pi[d];
        S[(size_t)(3 + si * 4 + 0) * SSTRIDE + sb] = (_Float16)ar;
        S[(size_t)(3 + si * 4 + 1) * SSTRIDE + sb] = (_Float16)ai;
        S[(size_t)(3 + si * 4 + 2) * SSTRIDE + sb] = (_Float16)br;
        S[(size_t)(3 + si * 4 + 3) * SSTRIDE + sb] = (_Float16)bi;
    }
}

// ---------------------------------------------------------------------------
// Kernel 2: build Toeplitz filter chunks in the WMMA A-matrix register layout.
// Contiguous k order chosen here (loadB matches): lane L holds row i = L%16; f16 element e (0..15)
// within the v16h maps to K = e + 16*(L/16).
// A[i,K] = w[32*chunk + K - i]  (0 outside [0,200]).
// ---------------------------------------------------------------------------
__global__ void prep_toeplitz(const float* __restrict__ wix,
                              const float* __restrict__ wic,
                              const float* __restrict__ fwr,
                              const float* __restrict__ fwi,
                              char* __restrict__ ws)
{
    int idx = blockIdx.x * blockDim.x + threadIdx.x;
    if (idx >= NFILT * NCHUNK * 32) return;
    int lane  = idx & 31;
    int fc    = idx >> 5;
    int chunk = fc % NCHUNK;
    int f     = fc / NCHUNK;
    int i = lane & 15, g = lane >> 4;

    _Float16* out = (_Float16*)(ws + WS_T_OFF) + (size_t)idx * 16;
    #pragma unroll
    for (int e = 0; e < 16; ++e) {
        int K = e + 16 * g;
        int t = 32 * chunk + K - i;
        float v = 0.f;
        if (t >= 0 && t < NTAPS) {
            if (f == 0)      v = wix[t];
            else if (f == 1) v = wic[t];
            else {
                int s = (f - 2) / 3, r = (f - 2) % 3;
                float wrv = fwr[s * NTAPS + t], wiv = fwi[s * NTAPS + t];
                v = (r == 0) ? wrv : (r == 1 ? wiv : -wiv);  // f16 A/B NEG illegal -> store -wi
            }
        }
        out[e] = (_Float16)v;
    }
}

// ---------------------------------------------------------------------------
// Kernel 3: WMMA conv + fused epilogue.  One wave per (16 output positions x
// 16 sequences) tile: 245 v_wmma_f32_16x16x32_f16 per tile, then the full
// SNSE epilogue in the D-matrix lane layout (row m = v + 8*(lane/16),
// col n = lane%16).
// ---------------------------------------------------------------------------
__device__ __forceinline__ v8f wmma_f16(v16h a, v16h b, v8f c) {
    return __builtin_amdgcn_wmma_f32_16x16x32_f16(false, a, false, b, (short)0, c, false, false);
}

__global__ void __launch_bounds__(128)
conv_main(const float* __restrict__ ti, const float* __restrict__ c00p,
          const float* __restrict__ wix, const float* __restrict__ wic,
          const char* __restrict__ ws, float* __restrict__ out)
{
    const int lane  = threadIdx.x & 31;
    const int gwave = blockIdx.x * (blockDim.x >> 5) + (threadIdx.x >> 5);
    const int G     = gwave & 1;          // sequence half: seqs 0..15 or 16..31
    const int n0    = (gwave >> 1) * 16;  // output-position tile base
    const int g     = lane >> 4;
    const int seq   = G * 16 + (lane & 15);   // this lane's B/D column

    const _Float16* T = (const _Float16*)(ws + WS_T_OFF);
    const _Float16* S = (const _Float16*)(ws + WS_S_OFF);
    const float2*   X = (const float2*)(ws + WS_X_OFF);

    // A operand: prebuilt in register layout -> one 32B v16h load per lane.
    auto loadA = [&](int f, int c) -> v16h {
        return *(const v16h*)(T + ((size_t)(f * NCHUNK + c) * 32 + lane) * 16);
    };
    // B operand (32x16, K x N): lane holds col n = lane%16, K = e + 16*(lane/16)
    // -> 16 contiguous f16 from the stream at window offset 32c + 16g.
    auto loadB = [&](int st, int c) -> v16h {
        return *(const v16h*)(S + (size_t)st * SSTRIDE + (size_t)seq * MPAD
                                + n0 + 32 * c + 16 * g);
    };

    // --- ixpm (over ps) and icixpm (over q.real/q.imag): 21 WMMAs ---
    v8f accI = {}, accQr = {}, accQi = {};
    #pragma unroll
    for (int ch = 0; ch < NCHUNK; ++ch) {
        v16h aI  = loadA(0, ch);
        v16h aC  = loadA(1, ch);
        v16h bP  = loadB(0, ch);
        v16h bQr = loadB(1, ch);
        v16h bQi = loadB(2, ch);
        accI  = wmma_f16(aI, bP,  accI);
        accQr = wmma_f16(aC, bQr, accQr);
        accQi = wmma_f16(aC, bQi, accQi);
    }

    // --- FWM: per shift, 4 complex-conv accumulators (56 WMMAs), combined
    //     immediately with the shifted x values. ---
    float fwr[8] = {0, 0, 0, 0, 0, 0, 0, 0};
    float fwi[8] = {0, 0, 0, 0, 0, 0, 0, 0};
    for (int si = 0; si < 4; ++si) {
        v8f ar = {}, ai = {}, br = {}, bi = {};
        const int fR = 2 + si * 3, fI = fR + 1, fN = fR + 2;   // wr, wi, -wi tables
        const int st = 3 + si * 4;                             // Ar, Ai, Br, Bi streams
        #pragma unroll
        for (int ch = 0; ch < NCHUNK; ++ch) {
            v16h awr = loadA(fR, ch), awi = loadA(fI, ch), anw = loadA(fN, ch);
            v16h bAr = loadB(st + 0, ch), bAi = loadB(st + 1, ch);
            v16h bBr = loadB(st + 2, ch), bBi = loadB(st + 3, ch);
            ar = wmma_f16(awr, bAr, ar);  ar = wmma_f16(anw, bAi, ar);  // conv(Ar,wr)-conv(Ai,wi)
            ai = wmma_f16(awi, bAr, ai);  ai = wmma_f16(awr, bAi, ai);  // conv(Ar,wi)+conv(Ai,wr)
            br = wmma_f16(awr, bBr, br);  br = wmma_f16(anw, bBi, br);
            bi = wmma_f16(awi, bBr, bi);  bi = wmma_f16(awr, bBi, bi);
        }
        const int s = (si < 2) ? (si - 2) : (si - 1);          // -2,-1,1,2
        #pragma unroll
        for (int v = 0; v < 8; ++v) {
            int p = n0 + v + 8 * g;
            float2 x1 = X[(size_t)seq * MM       + (p + TT - s)];  // x[p+100-s, c]
            float2 x2 = X[(size_t)(seq ^ 1) * MM + (p + TT - s)];  // x[p+100-s, 1-c]
            fwr[v] += x1.x * ar[v] - x1.y * ai[v] + x2.x * br[v] - x2.y * bi[v];
            fwi[v] += x1.x * ai[v] + x1.y * ar[v] + x2.x * bi[v] + x2.y * br[v];
        }
    }

    // --- fused epilogue ---
    const int   b     = seq >> 1;
    const int   c     = seq & 1;
    const float p10   = __expf(ti[b * 4] * 0.23025850929940457f);
    const float invsp = rsqrtf(p10 * 0.5f);                    // 1/sqrt(P)
    const float C00   = c00p[0];
    const float wI100 = wix[TT];
    const float wC100 = wic[TT];

    #pragma unroll
    for (int v = 0; v < 8; ++v) {
        int p = n0 + v + 8 * g;
        float2 xc = X[(size_t)seq * MM       + p + TT];        // center x_c
        float2 xo = X[(size_t)(seq ^ 1) * MM + p + TT];        // center x_{1-c}
        float pwc = xc.x * xc.x + xc.y * xc.y;
        float pwo = xo.x * xo.x + xo.y * xo.y;
        float psC = 2.f * pwc + pwo;
        float ixpm = 2.f * (accI[v] - wI100 * psC);            // Zcv: minus center tap
        float phi  = C00 * (pwc + pwo) + ixpm;                 // spm + ixpm
        float qrC = xc.x * xo.x + xc.y * xo.y;
        float qiC = xc.y * xo.x - xc.x * xo.y;
        float qcr = accQr[v] - wC100 * qrC;
        float qci = accQi[v] - wC100 * qiC;
        float icir = -(xo.x * qci + xo.y * qcr);               // i * xo * qc
        float icii =  (xo.x * qcr - xo.y * qci);
        float sn, cs;
        __sincosf(phi, &sn, &cs);
        float orr = (xc.x * cs - xc.y * sn + icir + fwr[v]) * invsp;
        float oii = (xc.x * sn + xc.y * cs + icii + fwi[v]) * invsp;
        if (p < MOUT) {
            size_t o = (((size_t)b * MOUT + p) * 2 + c) * 2;
            out[o]     = orr;
            out[o + 1] = oii;
        }
    }
}

// ---------------------------------------------------------------------------
extern "C" void kernel_launch(void* const* d_in, const int* in_sizes, int n_in,
                              void* d_out, int out_size, void* d_ws, size_t ws_size,
                              hipStream_t stream)
{
    (void)in_sizes; (void)n_in; (void)out_size; (void)ws_size;
    const float* xr  = (const float*)d_in[0];   // x_real  [16,40000,2]
    const float* xi  = (const float*)d_in[1];   // x_imag  [16,40000,2]
    const float* ti  = (const float*)d_in[2];   // task_info [16,4]
    const float* c00 = (const float*)d_in[3];   // C00 scalar
    const float* wix = (const float*)d_in[4];   // ixpm_w  [201]
    const float* wic = (const float*)d_in[5];   // icixpm_w[201]
    const float* fwr = (const float*)d_in[6];   // fwm_wr  [4,1,201]
    const float* fwi = (const float*)d_in[7];   // fwm_wi  [4,1,201]
    // d_in[8] ('i') unused: share=True -> module 0
    char*  ws  = (char*)d_ws;                   // needs ~59 MB scratch
    float* out = (float*)d_out;                 // [16,39800,2,2] f32

    prep_streams<<<(NSEQ * MPAD) / 256, 256, 0, stream>>>(xr, xi, ti, ws);
    prep_toeplitz<<<(NFILT * NCHUNK * 32 + 127) / 128, 128, 0, stream>>>(wix, wic, fwr, fwi, ws);
    conv_main<<<(NTILE * 2) / 4, 128, 0, stream>>>(ti, c00, wix, wic, ws, out);
}
